// TwoLayerNet_12945031430556
// MI455X (gfx1250) — compile-verified, matrix operand order checked by reading, not run on hardware
//
#include <hip/hip_runtime.h>
#include <hip/hip_bf16.h>

typedef __attribute__((ext_vector_type(16))) _Float16 v16h;
typedef __attribute__((ext_vector_type(4)))  _Float16 v4h;
typedef __attribute__((ext_vector_type(8)))  float    v8f;
typedef __attribute__((ext_vector_type(4)))  float    v4f;
typedef __attribute__((ext_vector_type(4)))  unsigned int v4u;

#define A_DIM 1024
#define Y_DIM 128
#define BATCH 16384
#define TM 64                 // rows per workgroup
#define AF_STRIDE 1032        // padded f16 row stride (16B aligned, bank-skewed)
#define SMEM_BYTES (TM * AF_STRIDE * 2 + TM * 2 * 4)
#define TOL 1e-3f

union Frag {
    v16h h;
    v4u  u[2];
};

// ---------------------------------------------------------------------------
// Kernel 1: WT[n][k] = (f16)( K[k, b]^2 * u[2k+b, y] ), n = b*128 + y.
// B-major layout so WMMA B fragments are contiguous 16B loads per lane.
// ---------------------------------------------------------------------------
__global__ __launch_bounds__(256) void build_wt(const float* __restrict__ K,
                                                const float* __restrict__ u,
                                                _Float16* __restrict__ WT) {
    int idx = blockIdx.x * 256 + threadIdx.x;   // 256*1024 total
    int k = idx & (A_DIM - 1);
    int n = idx >> 10;                          // 0..255
    int b = n >> 7;                             // 0: V0, 1: V1
    int y = n & (Y_DIM - 1);
    float kk = K[2 * k + b];
    float w = kk * kk * u[(size_t)(2 * k + b) * Y_DIM + y];
    WT[(size_t)n * A_DIM + k] = (_Float16)w;
}

// ---------------------------------------------------------------------------
// Kernel 2: fused fixed-point balancing (early-exit) + WMMA GEMM + combine.
// ---------------------------------------------------------------------------
__global__ __launch_bounds__(256) void fused_balance_gemm(
        const float* __restrict__ AT, const float* __restrict__ K,
        const float* __restrict__ BT, const float* __restrict__ bias,
        const _Float16* __restrict__ WT, float* __restrict__ out) {
    extern __shared__ char smem[];
    _Float16* afT = (_Float16*)smem;                       // [TM][AF_STRIDE]
    float*    sc  = (float*)(smem + TM * AF_STRIDE * 2);   // [TM][2] bf scales

    const int tid  = threadIdx.x;
    const int lane = tid & 31;
    const int wave = tid >> 5;            // 0..7
    const int row0 = blockIdx.x * TM;

    const float bt0 = BT[0];
    const float bt1 = BT[1];

    // ------------------ Phase 1: per-row fixed point ------------------
    // Lane owns elements a = j*128 + lane*4 + c (c=0..3), j=0..7: all loads
    // are 16B per lane, fully coalesced.
    v4f k0v[8], k1v[8];
#pragma unroll
    for (int j = 0; j < 8; ++j) {
        int a0 = j * 128 + lane * 4;
        v4f kA = *(const v4f*)(K + 2 * a0);       // [k(a0,0) k(a0,1) k(a0+1,0) k(a0+1,1)]
        v4f kB = *(const v4f*)(K + 2 * a0 + 4);
        k0v[j] = (v4f){kA[0] * kA[0], kA[2] * kA[2], kB[0] * kB[0], kB[2] * kB[2]};
        k1v[j] = (v4f){kA[1] * kA[1], kA[3] * kA[3], kB[1] * kB[1], kB[3] * kB[3]};
    }

    for (int rr = 0; rr < 8; ++rr) {
        const int row = wave * 8 + rr;
        const float* atp = AT + (size_t)(row0 + row) * A_DIM;
        v4f atv[8];
#pragma unroll
        for (int j = 0; j < 8; ++j)
            atv[j] = *(const v4f*)(atp + j * 128 + lane * 4);

        float bf0 = 0.f, bf1 = 0.f;
        float err = __builtin_inff();
        v4f afp[8];
#pragma unroll
        for (int j = 0; j < 8; ++j) afp[j] = (v4f){0.f, 0.f, 0.f, 0.f};

        // no-grad loop: while (i < 16 && err >= TOL), per-row error
#pragma unroll 1
        for (int it = 0; it < 16 && err >= TOL; ++it) {
            float s0 = 0.f, s1 = 0.f, em = 0.f;
#pragma unroll
            for (int j = 0; j < 8; ++j) {
                v4f at4 = atv[j], k04 = k0v[j], k14 = k1v[j], ap = afp[j], af4;
#pragma unroll
                for (int c = 0; c < 4; ++c) {
                    float d  = fmaf(bf0, k04[c], fmaf(bf1, k14[c], 1.0f));
                    float af = at4[c] * __builtin_amdgcn_rcpf(d);
                    s0 = fmaf(af, k04[c], s0);
                    s1 = fmaf(af, k14[c], s1);
                    em = fmaxf(em, fabsf(af - ap[c]));
                    af4[c] = af;
                }
                afp[j] = af4;
            }
#pragma unroll
            for (int m = 16; m >= 1; m >>= 1) {
                s0 += __shfl_xor(s0, m, 32);
                s1 += __shfl_xor(s1, m, 32);
                em = fmaxf(em, __shfl_xor(em, m, 32));
            }
            float nb0 = bt0 * __builtin_amdgcn_rcpf(s0 + 1.0f);
            float nb1 = bt1 * __builtin_amdgcn_rcpf(s1 + 1.0f);
            err = em + fmaxf(fabsf(nb0 - bf0), fabsf(nb1 - bf1));
            bf0 = nb0;
            bf1 = nb1;
        }

        // Final differentiable iteration: store af (f16), compute final bf.
        {
            float s0 = 0.f, s1 = 0.f;
#pragma unroll
            for (int j = 0; j < 8; ++j) {
                v4f at4 = atv[j], k04 = k0v[j], k14 = k1v[j];
                v4h h4;
#pragma unroll
                for (int c = 0; c < 4; ++c) {
                    float d  = fmaf(bf0, k04[c], fmaf(bf1, k14[c], 1.0f));
                    float af = at4[c] * __builtin_amdgcn_rcpf(d);
                    s0 = fmaf(af, k04[c], s0);
                    s1 = fmaf(af, k14[c], s1);
                    h4[c] = (_Float16)af;
                }
                *(v4h*)(afT + row * AF_STRIDE + j * 128 + lane * 4) = h4;
            }
#pragma unroll
            for (int m = 16; m >= 1; m >>= 1) {
                s0 += __shfl_xor(s0, m, 32);
                s1 += __shfl_xor(s1, m, 32);
            }
            bf0 = bt0 * __builtin_amdgcn_rcpf(s0 + 1.0f);
            bf1 = bt1 * __builtin_amdgcn_rcpf(s1 + 1.0f);
        }
        if (lane == 0) {
            sc[row * 2 + 0] = bf0;
            sc[row * 2 + 1] = bf1;
        }
    }
    __syncthreads();

    // ------------------ Phase 2: WMMA GEMM (M=64, K=1024, N=2x128) ------
    const int mtile = wave >> 1;          // 0..3  -> rows mtile*16..+15
    const int ng    = wave & 1;           // 0..1  -> y columns ng*64..+63
    const int h     = lane >> 4;          // lane half selects K sub-block
    const int l15   = lane & 15;

    v8f acc0[4], acc1[4];
#pragma unroll
    for (int t = 0; t < 4; ++t) {
        acc0[t] = (v8f){0.f, 0.f, 0.f, 0.f, 0.f, 0.f, 0.f, 0.f};
        acc1[t] = (v8f){0.f, 0.f, 0.f, 0.f, 0.f, 0.f, 0.f, 0.f};
    }

    const _Float16* aRow = afT + (mtile * 16 + l15) * AF_STRIDE + h * 8;

#pragma unroll 2
    for (int kk = 0; kk < A_DIM; kk += 32) {
        Frag A;
        A.u[0] = *(const v4u*)(aRow + kk);        // K = h*8 + 0..7
        A.u[1] = *(const v4u*)(aRow + kk + 16);   // K = h*8 + 16..23
#pragma unroll
        for (int t = 0; t < 4; ++t) {
            int n = (ng * 4 + t) * 16 + l15;      // output column 0..127
            const _Float16* b0p = WT + (size_t)n * A_DIM + kk + h * 8;
            const _Float16* b1p = b0p + (size_t)Y_DIM * A_DIM;  // V1 half
            Frag B0, B1;
            B0.u[0] = *(const v4u*)(b0p);
            B0.u[1] = *(const v4u*)(b0p + 16);
            B1.u[0] = *(const v4u*)(b1p);
            B1.u[1] = *(const v4u*)(b1p + 16);
            acc0[t] = __builtin_amdgcn_wmma_f32_16x16x32_f16(
                false, A.h, false, B0.h, (short)0, acc0[t], false, false);
            acc1[t] = __builtin_amdgcn_wmma_f32_16x16x32_f16(
                false, A.h, false, B1.h, (short)0, acc1[t], false, false);
        }
    }

    // ------------------ Combine: out = bf0*P0 + bf1*P1 + bias ----------
#pragma unroll
    for (int t = 0; t < 4; ++t) {
        int y = (ng * 4 + t) * 16 + l15;
        float bv = bias[y];
#pragma unroll
        for (int j = 0; j < 8; ++j) {
            int m   = j + h * 8;                  // C layout: M = j + 8*(lane>>4)
            int row = mtile * 16 + m;
            float b0 = sc[row * 2 + 0];
            float b1 = sc[row * 2 + 1];
            out[(size_t)(row0 + row) * Y_DIM + y] =
                fmaf(b0, acc0[t][j], fmaf(b1, acc1[t][j], bv));
        }
    }
}

extern "C" void kernel_launch(void* const* d_in, const int* in_sizes, int n_in,
                              void* d_out, int out_size, void* d_ws, size_t ws_size,
                              hipStream_t stream) {
    const float* AT   = (const float*)d_in[0];  // (16384, 1024)
    const float* K    = (const float*)d_in[1];  // (1024, 2)
    const float* BT   = (const float*)d_in[2];  // (2,)
    const float* u    = (const float*)d_in[3];  // (2048, 128)
    const float* bias = (const float*)d_in[4];  // (128,)
    float* out = (float*)d_out;                 // (16384, 128)

    _Float16* WT = (_Float16*)d_ws;             // 256*1024 f16 = 512 KB

    build_wt<<<(256 * A_DIM) / 256, 256, 0, stream>>>(K, u, WT);
    fused_balance_gemm<<<BATCH / TM, 256, SMEM_BYTES, stream>>>(
        AT, K, BT, bias, WT, out);
}
